// MultiHeadSelfAttention_89172110999728
// MI455X (gfx1250) — matrix operand order repaired and checked by reading, not compiled, not run on hardware
//
#include <hip/hip_runtime.h>

typedef __attribute__((ext_vector_type(8)))  __bf16 v8bf;
typedef __attribute__((ext_vector_type(16))) __bf16 v16bf;
typedef __attribute__((ext_vector_type(8)))  float  v8f;
typedef __attribute__((ext_vector_type(2)))  unsigned int v2u;

#define D_MODEL  1024
#define N_HEADS  16
#define HEAD_DIM 64
#define BATCH    4
#define SEQ      2048
#define ROWS     (BATCH * SEQ)   // 8192

__device__ __forceinline__ unsigned short f32_to_bf16(float f) {
  unsigned int u = __float_as_uint(f);
  u += 0x7FFFu + ((u >> 16) & 1u);   // round to nearest even
  return (unsigned short)(u >> 16);
}

__device__ __forceinline__ unsigned int pack2_bf16(float lo, float hi) {
  return (unsigned int)f32_to_bf16(lo) | ((unsigned int)f32_to_bf16(hi) << 16);
}

__device__ __forceinline__ v8f wmma_bf16(v16bf a, v16bf b, v8f c) {
  return __builtin_amdgcn_wmma_f32_16x16x32_bf16(
      false, a, false, b, (short)0, c, false, false);
}

__device__ __forceinline__ v16bf cat8(v8bf lo, v8bf hi) {
  return __builtin_shufflevector(lo, hi, 0,1,2,3,4,5,6,7,8,9,10,11,12,13,14,15);
}

// gfx1250 async global->LDS copy (ASYNCcnt-tracked DMA into LDS)
__device__ __forceinline__ void async_load_b128(unsigned ldsOff,
                                                const unsigned short* g) {
  asm volatile("global_load_async_to_lds_b128 %0, %1, off"
               :: "v"(ldsOff), "v"(g) : "memory");
}

// ---------------------------------------------------------------- convert (vectorized: 4 f32 -> 4 bf16)
__global__ void convert_f32_bf16(const float* __restrict__ in,
                                 unsigned short* __restrict__ out, int n4) {
  int i = blockIdx.x * blockDim.x + threadIdx.x;
  int stride = gridDim.x * blockDim.x;
  for (; i < n4; i += stride) {
    float4 v = ((const float4*)in)[i];
    v2u p;
    p.x = pack2_bf16(v.x, v.y);
    p.y = pack2_bf16(v.z, v.w);
    ((v2u*)out)[i] = p;
  }
}

// ---------------------------------------------------------------- V transpose: [b,s,h*64+d] -> [b,h,d,s]
__global__ void transpose_v(const unsigned short* __restrict__ Vb,
                            unsigned short* __restrict__ Vt) {
  const int total = BATCH * N_HEADS * HEAD_DIM * SEQ;
  int i = blockIdx.x * blockDim.x + threadIdx.x;
  int stride = gridDim.x * blockDim.x;
  for (; i < total; i += stride) {
    int s = i & (SEQ - 1);
    int d = (i >> 11) & (HEAD_DIM - 1);
    int h = (i >> 17) & (N_HEADS - 1);
    int b = i >> 21;
    Vt[i] = Vb[(size_t)(b * SEQ + s) * D_MODEL + h * HEAD_DIM + d];
  }
}

// ---------------------------------------------------------------- NT GEMM: C[M,N] = outScale * A[M,K] * B[N,K]^T
// B tile (64 rows x 64 k, shared by all 8 waves) staged into LDS via async DMA,
// double-buffered so the copy of tile k+1 overlaps WMMAs on tile k.
template <bool BF16_OUT>
__global__ void gemm_nt_wmma(const unsigned short* __restrict__ A,
                             const unsigned short* __restrict__ Bw,
                             void* __restrict__ Cout, int M, int N, int K,
                             float outScale) {
  __shared__ unsigned short bTile[2][64 * 64];   // 2 x 8KB

  const int wave = threadIdx.x >> 5;
  const int lane = threadIdx.x & 31;
  const int hf   = lane >> 4;
  const int ln   = lane & 15;
  const int t    = threadIdx.x;

  const int mBlocks = M >> 7;                 // 128 rows per block
  const int bm = blockIdx.x % mBlocks;
  const int bn = blockIdx.x / mBlocks;
  const int mBase = bm * 128 + wave * 16;
  const int nBase = bn * 64;

  v8f acc0 = {}, acc1 = {}, acc2 = {}, acc3 = {};

  const unsigned short* aRow = A + (size_t)(mBase + ln) * K;

  // stage B[nBase..+63][k0..+63] into bTile[buf]: 8KB = 512 x b128, 2 per thread
  auto stage = [&](int k0, int buf) {
#pragma unroll
    for (int c = t; c < 512; c += 256) {
      const int row = c >> 3;
      const int col = (c & 7) * 8;
      unsigned ldsOff = (unsigned)(size_t)&bTile[buf][row * 64 + col];
      const unsigned short* g = Bw + (size_t)(nBase + row) * K + k0 + col;
      async_load_b128(ldsOff, g);
    }
  };

  stage(0, 0);
  int buf = 0;
  for (int k0 = 0; k0 < K; k0 += 64) {
    if (k0 + 64 < K) {
      stage(k0 + 64, buf ^ 1);
      // ASYNC ops complete in order: <=2 outstanding drains the current buffer
      asm volatile("s_wait_asynccnt 0x2" ::: "memory");
    } else {
      asm volatile("s_wait_asynccnt 0x0" ::: "memory");
    }
    __syncthreads();

    const unsigned short* bt = &bTile[buf][0];
#pragma unroll
    for (int ks = 0; ks < 2; ++ks) {
      const int k = k0 + ks * 32;
      v16bf af = *(const v16bf*)(aRow + k + hf * 16);
      const unsigned short* bB = bt + ln * 64 + ks * 32 + hf * 16;
      v16bf b0 = *(const v16bf*)(bB);
      v16bf b1 = *(const v16bf*)(bB + 16 * 64);
      v16bf b2 = *(const v16bf*)(bB + 32 * 64);
      v16bf b3 = *(const v16bf*)(bB + 48 * 64);
      acc0 = wmma_bf16(af, b0, acc0);
      acc1 = wmma_bf16(af, b1, acc1);
      acc2 = wmma_bf16(af, b2, acc2);
      acc3 = wmma_bf16(af, b3, acc3);
    }
    __syncthreads();
    buf ^= 1;
  }

#pragma unroll
  for (int r = 0; r < 8; ++r) {
    int row = mBase + r + 8 * hf;
    size_t base = (size_t)row * N + nBase + ln;
    if (BF16_OUT) {
      unsigned short* C = (unsigned short*)Cout;
      C[base]      = f32_to_bf16(acc0[r] * outScale);
      C[base + 16] = f32_to_bf16(acc1[r] * outScale);
      C[base + 32] = f32_to_bf16(acc2[r] * outScale);
      C[base + 48] = f32_to_bf16(acc3[r] * outScale);
    } else {
      float* C = (float*)Cout;
      C[base]      = acc0[r] * outScale;
      C[base + 16] = acc1[r] * outScale;
      C[base + 32] = acc2[r] * outScale;
      C[base + 48] = acc3[r] * outScale;
    }
  }
}

// ---------------------------------------------------------------- flash attention (transposed pipeline)
// S^T = K*Q^T  ->  lane = query column; per-lane scalar softmax state.
// O^T = V^T*P^T -> lane = query column; scalar alpha/1/l rescale.
template <bool MASKED>
__device__ __forceinline__ void attn_kblock(
    int kb, int qBase, int ln, int hf,
    const unsigned short* __restrict__ kRowBase,
    const unsigned short* __restrict__ vRowBase,
    const v16bf& qf0, const v16bf& qf1,
    unsigned short* __restrict__ myP,
    float& mrun, float& lrun,
    v8f& o0, v8f& o1, v8f& o2, v8f& o3) {
  // ---- S^T tiles (keys x queries); A = K rows, B = Q^T
  v8f st0 = {}, st1 = {};
  const unsigned short* kr0 = kRowBase + (size_t)(kb * 32) * D_MODEL;
  const unsigned short* kr1 = kr0 + (size_t)16 * D_MODEL;
  {
    v16bf ka0 = cat8(*(const v8bf*)(kr0 + hf * 8),      *(const v8bf*)(kr0 + 16 + hf * 8));
    v16bf ka1 = cat8(*(const v8bf*)(kr1 + hf * 8),      *(const v8bf*)(kr1 + 16 + hf * 8));
    st0 = wmma_bf16(ka0, qf0, st0);
    st1 = wmma_bf16(ka1, qf0, st1);
    ka0 = cat8(*(const v8bf*)(kr0 + 32 + hf * 8), *(const v8bf*)(kr0 + 48 + hf * 8));
    ka1 = cat8(*(const v8bf*)(kr1 + 32 + hf * 8), *(const v8bf*)(kr1 + 48 + hf * 8));
    st0 = wmma_bf16(ka0, qf1, st0);
    st1 = wmma_bf16(ka1, qf1, st1);
  }

  // ---- per-lane softmax over this block (16 scores of query qBase+ln)
  float s0[8], s1[8];
  const int q = qBase + ln;
#pragma unroll
  for (int r = 0; r < 8; ++r) {
    s0[r] = st0[r];
    s1[r] = st1[r];
    if (MASKED) {
      const int key = kb * 32 + r + 8 * hf;
      if (key > q)      s0[r] = -1.0e9f;
      if (key + 16 > q) s1[r] = -1.0e9f;
    }
  }
  float t = s0[0];
#pragma unroll
  for (int r = 1; r < 8; ++r) t = fmaxf(t, s0[r]);
#pragma unroll
  for (int r = 0; r < 8; ++r) t = fmaxf(t, s1[r]);
  t = fmaxf(t, __shfl_xor(t, 16, 32));           // combine partner half (other 16 keys)
  const float mnew = fmaxf(mrun, t);
  const float alpha = __expf(mrun - mnew);
  mrun = mnew;

  float sum = 0.0f;
#pragma unroll
  for (int r = 0; r < 8; ++r) {
    s0[r] = __expf(s0[r] - mnew);
    s1[r] = __expf(s1[r] - mnew);
    sum += s0[r] + s1[r];
  }
  sum += __shfl_xor(sum, 16, 32);
  lrun = lrun * alpha + sum;

  // ---- rescale O^T accumulators (alpha is per-query -> per-lane scalar)
#pragma unroll
  for (int r = 0; r < 8; ++r) {
    o0[r] *= alpha; o1[r] *= alpha; o2[r] *= alpha; o3[r] *= alpha;
  }

  // ---- P^T to LDS: layout [query][key0..31]; lane holds consecutive key pairs
  unsigned int* ldsW = (unsigned int*)myP;
#pragma unroll
  for (int i = 0; i < 4; ++i) {
    ldsW[(ln * 32 + hf * 8) / 2 + i]      = pack2_bf16(s0[2 * i], s0[2 * i + 1]);
    ldsW[(ln * 32 + 16 + hf * 8) / 2 + i] = pack2_bf16(s1[2 * i], s1[2 * i + 1]);
  }

  // ---- single B-fragment of P^T shared by all 4 d-tiles
  v16bf pf = cat8(*(const v8bf*)(myP + ln * 32 + hf * 8), *(const v8bf*)(myP + ln * 32 + 16 + hf * 8));

  // ---- O^T += V^T * P^T   (A = V^T rows from Vt, contiguous in s)
  const unsigned short* vr = vRowBase + kb * 32;
  v16bf va;
  va = cat8(*(const v8bf*)(vr + hf * 8), *(const v8bf*)(vr + 16 + hf * 8));
  o0 = wmma_bf16(va, pf, o0);
  vr += (size_t)16 * SEQ;
  va = cat8(*(const v8bf*)(vr + hf * 8), *(const v8bf*)(vr + 16 + hf * 8));
  o1 = wmma_bf16(va, pf, o1);
  vr += (size_t)16 * SEQ;
  va = cat8(*(const v8bf*)(vr + hf * 8), *(const v8bf*)(vr + 16 + hf * 8));
  o2 = wmma_bf16(va, pf, o2);
  vr += (size_t)16 * SEQ;
  va = cat8(*(const v8bf*)(vr + hf * 8), *(const v8bf*)(vr + 16 + hf * 8));
  o3 = wmma_bf16(va, pf, o3);
}

__global__ void attn_flash_wmma(const unsigned short* __restrict__ Qb,
                                const unsigned short* __restrict__ Kb,
                                const unsigned short* __restrict__ Vt,
                                unsigned short* __restrict__ Ob) {
  __shared__ unsigned short ldsP[8][16 * 32];   // per-wave 16x32 P^T tile

  const int wave = threadIdx.x >> 5;
  const int lane = threadIdx.x & 31;
  const int hf   = lane >> 4;
  const int ln   = lane & 15;

  const int gw = blockIdx.x * 8 + wave;         // 0..8191
  const int qt = gw & 127;
  const int h  = (gw >> 7) & (N_HEADS - 1);
  const int b  = gw >> 11;
  const int qBase = qt * 16;

  // Q^T B-fragments (d x query): lane n = query, d = 8*hf+s (s<8) / 16+8*hf+(s-8) (s>=8), +32 for qf1
  const unsigned short* qRow =
      Qb + (size_t)(b * SEQ + qBase + ln) * D_MODEL + h * HEAD_DIM;
  const v16bf qf0 = cat8(*(const v8bf*)(qRow + hf * 8), *(const v8bf*)(qRow + 16 + hf * 8));
  const v16bf qf1 = cat8(*(const v8bf*)(qRow + 32 + hf * 8), *(const v8bf*)(qRow + 48 + hf * 8));

  const unsigned short* kRowBase =
      Kb + (size_t)(b * SEQ + ln) * D_MODEL + h * HEAD_DIM;
  const unsigned short* vRowBase =
      Vt + ((size_t)((b * N_HEADS + h) * HEAD_DIM) + ln) * SEQ;

  unsigned short* myP = &ldsP[wave][0];

  float mrun = -3.0e38f, lrun = 0.0f;
  v8f o0 = {}, o1 = {}, o2 = {}, o3 = {};

  const int nFull = qBase >> 5;                 // fully-unmasked 32-key blocks
  for (int kb = 0; kb < nFull; ++kb)
    attn_kblock<false>(kb, qBase, ln, hf, kRowBase, vRowBase, qf0, qf1,
                       myP, mrun, lrun, o0, o1, o2, o3);
  attn_kblock<true>(nFull, qBase, ln, hf, kRowBase, vRowBase, qf0, qf1,
                    myP, mrun, lrun, o0, o1, o2, o3);

  // ---- normalize, store O^T (lane = query; VGPR r -> d = tile*16 + r + 8*hf)
  const float inv = 1.0f / lrun;
  unsigned int* outRow = (unsigned int*)
      (Ob + (size_t)(b * SEQ + qBase + ln) * D_MODEL + h * HEAD_DIM);
#pragma unroll
  for (int i = 0; i < 4; ++i) {
    outRow[(0 * 16 + 8 * hf) / 2 + i] = pack2_bf16(o0[2 * i] * inv, o0[2 * i + 1] * inv);
    outRow[(1 * 16 + 8 * hf) / 2 + i] = pack2_bf16(o1[2 * i] * inv, o1[2 * i + 1] * inv);
    outRow[(2 * 16 + 8 * hf) / 2 + i] = pack2_bf16(o2[2 * i] * inv, o2[2 * i + 1] * inv);
    outRow[(3 * 16 + 8 * hf) / 2 + i] = pack2_bf16(o3[2 * i] * inv, o3[2 * i + 1] * inv);
  }
}

// ---------------------------------------------------------------- launch
extern "C" void kernel_launch(void* const* d_in, const int* in_sizes, int n_in,
                              void* d_out, int out_size, void* d_ws, size_t ws_size,
                              hipStream_t stream) {
  (void)in_sizes; (void)n_in; (void)out_size; (void)ws_size;

  const float* x  = (const float*)d_in[0];
  const float* wq = (const float*)d_in[1];
  const float* wk = (const float*)d_in[2];
  const float* wv = (const float*)d_in[3];
  const float* wo = (const float*)d_in[4];

  const size_t ACT = (size_t)ROWS * D_MODEL;       // 8M elements
  const size_t WSZ = (size_t)D_MODEL * D_MODEL;    // 1M elements
  char* ws = (char*)d_ws;
  size_t off = 0;
  unsigned short* Xb   = (unsigned short*)(ws + off); off += ACT * 2;
  unsigned short* Wq   = (unsigned short*)(ws + off); off += WSZ * 2;
  unsigned short* Wk   = (unsigned short*)(ws + off); off += WSZ * 2;
  unsigned short* Wv   = (unsigned short*)(ws + off); off += WSZ * 2;
  unsigned short* Wo   = (unsigned short*)(ws + off); off += WSZ * 2;
  unsigned short* Qb   = (unsigned short*)(ws + off); off += ACT * 2;
  unsigned short* Kbuf = (unsigned short*)(ws + off); off += ACT * 2;
  unsigned short* Vb   = (unsigned short*)(ws + off); off += ACT * 2;
  unsigned short* Vt   = (unsigned short*)(ws + off); off += ACT * 2;
  unsigned short* Ob   = (unsigned short*)(ws + off); off += ACT * 2;

  // 1) f32 -> bf16 (4 elems/thread)
  convert_f32_bf16<<<2048, 256, 0, stream>>>(x,  Xb, (int)(ACT / 4));
  convert_f32_bf16<<<512,  256, 0, stream>>>(wq, Wq, (int)(WSZ / 4));
  convert_f32_bf16<<<512,  256, 0, stream>>>(wk, Wk, (int)(WSZ / 4));
  convert_f32_bf16<<<512,  256, 0, stream>>>(wv, Wv, (int)(WSZ / 4));
  convert_f32_bf16<<<512,  256, 0, stream>>>(wo, Wo, (int)(WSZ / 4));

  // 2) projections; 1/sqrt(HEAD_DIM) folded into Q
  gemm_nt_wmma<true><<<1024, 256, 0, stream>>>(Xb, Wq, Qb,   ROWS, D_MODEL, D_MODEL, 0.125f);
  gemm_nt_wmma<true><<<1024, 256, 0, stream>>>(Xb, Wk, Kbuf, ROWS, D_MODEL, D_MODEL, 1.0f);
  gemm_nt_wmma<true><<<1024, 256, 0, stream>>>(Xb, Wv, Vb,   ROWS, D_MODEL, D_MODEL, 1.0f);

  // 3) V -> per-head transposed layout [b,h,d,s]
  transpose_v<<<8192, 256, 0, stream>>>(Vb, Vt);

  // 4) causal flash attention (8192 waves = 1024 blocks of 8 waves)
  attn_flash_wmma<<<1024, 256, 0, stream>>>(Qb, Kbuf, Vt, Ob);

  // 5) output projection -> f32 d_out
  gemm_nt_wmma<false><<<1024, 256, 0, stream>>>(Ob, Wo, d_out, ROWS, D_MODEL, D_MODEL, 1.0f);
}
